// WindowAttention_69217692942500
// MI455X (gfx1250) — matrix operand order repaired and checked by reading, not compiled, not run on hardware
//
#include <hip/hip_runtime.h>
#include <stdint.h>

// ---------------------------------------------------------------------------
// Types for CDNA5 WMMA (wave32): D[16x16 f32] = A[16x32 bf16] * B[32x16 bf16] + C
// ---------------------------------------------------------------------------
typedef __attribute__((ext_vector_type(16))) __bf16          v16bf;
typedef __attribute__((ext_vector_type(2)))  __bf16          v2bf;
typedef __attribute__((ext_vector_type(8)))  float           v8f;
typedef __attribute__((ext_vector_type(16))) unsigned short  us16;
typedef __attribute__((ext_vector_type(8)))  unsigned short  us8;
typedef __attribute__((ext_vector_type(4)))  unsigned int    u32x4;
typedef __attribute__((ext_vector_type(4)))  float           f32x4;
typedef __attribute__((ext_vector_type(4)))  int             v4i;
typedef __attribute__((address_space(1)))    v4i             as1_v4i;  // global
typedef __attribute__((address_space(3)))    v4i             as3_v4i;  // LDS

#if __has_builtin(__builtin_amdgcn_global_load_async_to_lds_b128) && \
    __has_builtin(__builtin_amdgcn_s_wait_asynccnt)
#define HAVE_ASYNC_LDS 1
#else
#define HAVE_ASYNC_LDS 0
#endif

// Packed f32x2 -> bf16x2. Prefer the packed-cvt builtin; otherwise rely on the
// target's native __bf16 truncation (gfx1250 is a bf16-first ISA).
__device__ __forceinline__ unsigned int pack_bf16x2(float lo, float hi) {
#if __has_builtin(__builtin_amdgcn_cvt_pk_bf16_f32)
  v2bf r = __builtin_amdgcn_cvt_pk_bf16_f32(lo, hi);
#else
  v2bf r = {(__bf16)lo, (__bf16)hi};
#endif
  return __builtin_bit_cast(unsigned int, r);
}
// Single f32 -> bf16 (for scattered b16 stores).
__device__ __forceinline__ unsigned short bf16lo(float v) {
  return __builtin_bit_cast(unsigned short, (__bf16)v);
}

// Async DMA: global -> LDS, 16B per lane, tracked by ASYNCcnt.
__device__ __forceinline__ void async_copy16(const void* g, void* l) {
#if HAVE_ASYNC_LDS
  __builtin_amdgcn_global_load_async_to_lds_b128(
      (as1_v4i*)(g), (as3_v4i*)(l), 0, 0);
#else
  *(us8*)l = *(const us8*)g;
#endif
}
__device__ __forceinline__ void async_join() {
#if HAVE_ASYNC_LDS
  __builtin_amdgcn_s_wait_asynccnt(0);
#endif
}

__device__ __forceinline__ us16 cat8(us8 lo, us8 hi) {
  return __builtin_shufflevector(lo, hi, 0,1,2,3,4,5,6,7,8,9,10,11,12,13,14,15);
}

__device__ __forceinline__ v8f vzero8() {
  v8f z;
#pragma unroll
  for (int i = 0; i < 8; ++i) z[i] = 0.f;
  return z;
}

__device__ __forceinline__ v8f wmma_bf16(us16 a, us16 b, v8f c) {
  return __builtin_amdgcn_wmma_f32_16x16x32_bf16(
      /*neg_a=*/false, __builtin_bit_cast(v16bf, a),
      /*neg_b=*/false, __builtin_bit_cast(v16bf, b),
      /*c_mod=*/(short)0, c, /*reuse_a=*/false, /*reuse_b=*/false);
}

// A-fragment (16x32, bf16): lane L = row (L&15); elements 0..15 = K base..base+15,
// base = (L>>4)*16
__device__ __forceinline__ us16 frag_a_lds(const unsigned short* base_row, int lane) {
  int b = (lane >> 4) * 16;
  us8 lo = *(const us8*)(base_row + b);
  us8 hi = *(const us8*)(base_row + b + 8);
  return cat8(lo, hi);
}
// B-fragment (32x16, bf16): lane L = column (L&15); elements 0..15 = K kg..kg+15,
// kg = (L>>4)*16. base_row points at the (column-major source) row for this lane.
__device__ __forceinline__ us16 frag_b_lds(const unsigned short* base_row, int lane) {
  int kg = (lane >> 4) * 16;
  us8 lo = *(const us8*)(base_row + kg);
  us8 hi = *(const us8*)(base_row + kg + 8);
  return cat8(lo, hi);
}

// ---------------------------------------------------------------------------
// Problem constants
// ---------------------------------------------------------------------------
#define DIMC     256
#define HEADS    8
#define HEAD_DIM 32
#define NTOK     64
#define NWIN     64
#define BTOT     1024            // window-batches
#define MROWS    (BTOT * NTOK)   // 65536 token rows

// GEMM tiling
#define BM 128
#define BN 128
#define BK 32
#define APITCH 40                // bf16 elems per LDS row (80B -> conflict-free b128)

// ---------------------------------------------------------------------------
// Weight f32 -> bf16 conversion
// ---------------------------------------------------------------------------
__global__ void cvt_bf16_kernel(const float* __restrict__ src,
                                unsigned short* __restrict__ dst, int n) {
  int i = 2 * (blockIdx.x * blockDim.x + threadIdx.x);
  if (i < n)
    *(unsigned int*)(dst + i) = pack_bf16x2(src[i], src[i + 1]);
}

// ---------------------------------------------------------------------------
// Tiled GEMM: out[m, o] = (sum_c A[m,c] * W[o,c]) + bias[o], all K=256.
// MODE 0: A f32, epilogue -> bf16 [B][H][N][hd]          (q / k), * outScale
// MODE 1: A f32, epilogue -> bf16 [B][H][hd][N] (transposed v / vscale / vshift)
// MODE 2: A bf16, epilogue -> f32 row-major [m][o]       (final projections)
// Block: 256 threads = 8 waves; wave (mw,nw) computes 32x64 via 8 WMMA / K-step.
// bf16 tiles (B always; A in MODE 2) are staged with async global->LDS DMA.
// ---------------------------------------------------------------------------
template <int MODE>
__global__ __launch_bounds__(256) void gemm_kernel(
    const void* __restrict__ Aptr, const unsigned short* __restrict__ Bw,
    const float* __restrict__ bias, void* __restrict__ outp, float outScale) {
  __shared__ unsigned short ldsA[2][BM * APITCH];
  __shared__ unsigned short ldsB[2][BN * APITCH];

  const int tid  = threadIdx.x;
  const int lane = tid & 31;
  const int wave = tid >> 5;
  const int mw   = wave & 3;            // 4 waves along M
  const int nw   = wave >> 2;           // 2 waves along N
  const int blockM = blockIdx.x * BM;
  const int blockN = blockIdx.y * BN;

  v8f acc[2][4];
#pragma unroll
  for (int i = 0; i < 2; ++i)
#pragma unroll
    for (int j = 0; j < 4; ++j) acc[i][j] = vzero8();

  auto stage = [&](int buf, int ks) {
    const int r = tid & 127;
    if (tid < 128) {                    // stage A rows
      if (MODE == 2) {                  // already bf16: async DMA to LDS
        const unsigned short* src =
            (const unsigned short*)Aptr + (size_t)(blockM + r) * DIMC + ks * BK;
#pragma unroll
        for (int i = 0; i < 4; ++i)
          async_copy16(src + i * 8, &ldsA[buf][r * APITCH + i * 8]);
      } else {                          // f32: load, packed-convert, ds_store
        const f32x4* src = (const f32x4*)((const float*)Aptr +
                                          (size_t)(blockM + r) * DIMC + ks * BK);
        u32x4* dst = (u32x4*)&ldsA[buf][r * APITCH];
#pragma unroll
        for (int i = 0; i < 4; ++i) {
          f32x4 x = src[2 * i], y = src[2 * i + 1];
          u32x4 o;
          o[0] = pack_bf16x2(x[0], x[1]);
          o[1] = pack_bf16x2(x[2], x[3]);
          o[2] = pack_bf16x2(y[0], y[1]);
          o[3] = pack_bf16x2(y[2], y[3]);
          dst[i] = o;
        }
      }
    } else {                            // stage B rows (bf16 weights): async DMA
      const unsigned short* src = Bw + (size_t)(blockN + r) * DIMC + ks * BK;
#pragma unroll
      for (int i = 0; i < 4; ++i)
        async_copy16(src + i * 8, &ldsB[buf][r * APITCH + i * 8]);
    }
  };

  stage(0, 0);
  async_join();
  __syncthreads();

#pragma unroll 1
  for (int ks = 0; ks < DIMC / BK; ++ks) {
    const int cur = ks & 1;
    if (ks + 1 < DIMC / BK) stage(cur ^ 1, ks + 1);

    us16 afr[2], bfr[4];
#pragma unroll
    for (int mt = 0; mt < 2; ++mt)
      afr[mt] = frag_a_lds(&ldsA[cur][(mw * 32 + mt * 16 + (lane & 15)) * APITCH], lane);
#pragma unroll
    for (int nt = 0; nt < 4; ++nt)
      bfr[nt] = frag_b_lds(&ldsB[cur][(nw * 64 + nt * 16 + (lane & 15)) * APITCH], lane);
#pragma unroll
    for (int mt = 0; mt < 2; ++mt)
#pragma unroll
      for (int nt = 0; nt < 4; ++nt) acc[mt][nt] = wmma_bf16(afr[mt], bfr[nt], acc[mt][nt]);

    async_join();                        // async stage of next buffer complete
    __syncthreads();
  }

  // Epilogue. C layout: VGPR j -> M = j (lanes 0-15) / 8+j (lanes 16-31), N = lane&15.
  const int mh = (lane >> 4) * 8;
#pragma unroll
  for (int mt = 0; mt < 2; ++mt) {
#pragma unroll
    for (int nt = 0; nt < 4; ++nt) {
      const int o = blockN + nw * 64 + nt * 16 + (lane & 15);
      const float bv = bias[o];
#pragma unroll
      for (int j = 0; j < 8; ++j) {
        const int grow = blockM + mw * 32 + mt * 16 + j + mh;
        float v = (acc[mt][nt][j] + bv) * outScale;
        if (MODE == 2) {
          ((float*)outp)[(size_t)grow * DIMC + o] = v;
        } else {
          const int b = grow >> 6, nn = grow & 63, h = o >> 5, d = o & 31;
          size_t idx;
          if (MODE == 0) idx = ((size_t)((b * HEADS + h) * NTOK + nn)) * HEAD_DIM + d;
          else           idx = ((size_t)((b * HEADS + h) * HEAD_DIM + d)) * NTOK + nn;
          ((unsigned short*)outp)[idx] = bf16lo(v);
        }
      }
    }
  }
}

// ---------------------------------------------------------------------------
// Attention: one block = one (window b, head h). 128 threads = 4 waves.
// scores = q*scale @ k^T (K=32, one WMMA per 16x16 tile) + rel_bias + mask,
// softmax, then probs @ {v, vscale, vshift} (K=64, bf16 probs from LDS).
// ---------------------------------------------------------------------------
__global__ __launch_bounds__(128) void attn_kernel(
    const unsigned short* __restrict__ qb, const unsigned short* __restrict__ kb,
    const unsigned short* __restrict__ vT, const unsigned short* __restrict__ vsT,
    const unsigned short* __restrict__ vhT, const float* __restrict__ mask,
    const float* __restrict__ bias_table, const int* __restrict__ rel_idx,
    unsigned short* __restrict__ O, unsigned short* __restrict__ Os,
    unsigned short* __restrict__ Oh) {
  __shared__ float          sAttn[NTOK][NTOK + 2];
  __shared__ unsigned short sProb[NTOK][NTOK + 8];   // pitch 72 (144B, 16B-aligned)

  const int tid = threadIdx.x, lane = tid & 31, wave = tid >> 5;
  const int bh = blockIdx.x;
  const int b = bh >> 3, h = bh & 7;
  const size_t headOff = (size_t)(b * HEADS + h) * (NTOK * HEAD_DIM);
  const int mh = (lane >> 4) * 8;

  // ---- scores = q @ k^T -------------------------------------------------
  {
    const int row = wave * 16 + (lane & 15);
    const unsigned short* qrow = qb + headOff + row * HEAD_DIM;
    us16 afr = frag_a_lds(qrow, lane);          // q rows contiguous along d
    const int kg = (lane >> 4) * 16;
#pragma unroll
    for (int t = 0; t < 4; ++t) {
      const int mcol = t * 16 + (lane & 15);
      const unsigned short* krow = kb + headOff + mcol * HEAD_DIM + kg;
      us16 bfr = cat8(*(const us8*)krow, *(const us8*)(krow + 8));
      v8f c = wmma_bf16(afr, bfr, vzero8());
#pragma unroll
      for (int j = 0; j < 8; ++j)
        sAttn[wave * 16 + j + mh][t * 16 + (lane & 15)] = c[j];
    }
  }
  __syncthreads();

  // ---- softmax with relative-position bias + shifted-window mask --------
  if (tid < NTOK) {
    const int n = tid, w = b & (NWIN - 1);
    const float* mrow = mask + ((size_t)w * NTOK + n) * NTOK;
    const int* rrow = rel_idx + n * NTOK;
    float mx = -3.0e38f;
#pragma unroll 4
    for (int m = 0; m < NTOK; ++m) {
      float s = sAttn[n][m] + bias_table[rrow[m] * HEADS + h] + mrow[m];
      sAttn[n][m] = s;
      mx = fmaxf(mx, s);
    }
    float sum = 0.f;
#pragma unroll 4
    for (int m = 0; m < NTOK; ++m) {
      float e = __expf(sAttn[n][m] - mx);
      sAttn[n][m] = e;
      sum += e;
    }
    const float inv = 1.0f / sum;
    unsigned int* prow = (unsigned int*)&sProb[n][0];
#pragma unroll 4
    for (int m = 0; m < NTOK; m += 2)
      prow[m >> 1] = pack_bf16x2(sAttn[n][m] * inv, sAttn[n][m + 1] * inv);
  }
  __syncthreads();

  // ---- out = probs @ {v, vs, vh} (values stored transposed [hd][N]) -----
  const unsigned short* srcs[3] = {vT, vsT, vhT};
  unsigned short* dsts[3] = {O, Os, Oh};
  v8f acc[3][2];
#pragma unroll
  for (int j = 0; j < 3; ++j)
#pragma unroll
    for (int t = 0; t < 2; ++t) acc[j][t] = vzero8();

#pragma unroll
  for (int ksb = 0; ksb < 2; ++ksb) {           // K = 64 -> two steps of 32
    const int row = wave * 16 + (lane & 15);
    us16 afr = frag_a_lds(&sProb[row][ksb * 32], lane);
    const int kg = ksb * 32 + (lane >> 4) * 16;
#pragma unroll
    for (int t = 0; t < 2; ++t) {
      const int d = t * 16 + (lane & 15);
      const size_t vrow = ((size_t)(b * HEADS + h) * HEAD_DIM + d) * NTOK + kg;
#pragma unroll
      for (int j = 0; j < 3; ++j) {
        const unsigned short* p = srcs[j] + vrow;
        us16 bfr = cat8(*(const us8*)p, *(const us8*)(p + 8));
        acc[j][t] = wmma_bf16(afr, bfr, acc[j][t]);
      }
    }
  }

  // Epilogue: O[(b*64+m), h*32+d] bf16 row-major for final projections.
#pragma unroll
  for (int j = 0; j < 3; ++j)
#pragma unroll
    for (int t = 0; t < 2; ++t)
#pragma unroll
      for (int jj = 0; jj < 8; ++jj) {
        const int m = wave * 16 + jj + mh;
        const int d = t * 16 + (lane & 15);
        dsts[j][(size_t)(b * NTOK + m) * DIMC + h * HEAD_DIM + d] =
            bf16lo(acc[j][t][jj]);
      }
}

// ---------------------------------------------------------------------------
// Host launcher
// ---------------------------------------------------------------------------
extern "C" void kernel_launch(void* const* d_in, const int* in_sizes, int n_in,
                              void* d_out, int out_size, void* d_ws, size_t ws_size,
                              hipStream_t stream) {
  (void)in_sizes; (void)n_in; (void)out_size; (void)ws_size;

  const float* mask       = (const float*)d_in[5];
  const float* bias_table = (const float*)d_in[22];
  const int*   rel_idx    = (const int*)d_in[23];

  // Workspace carve-out (all bf16 intermediates); total ~257 MB.
  char* ws = (char*)d_ws;
  const size_t WB  = (size_t)DIMC * DIMC * sizeof(unsigned short);   // 128 KB / matrix
  const size_t TB  = (size_t)MROWS * DIMC * sizeof(unsigned short);  // 32 MB / tensor
  unsigned short* Wbf = (unsigned short*)ws;            size_t off = 8 * WB;
  unsigned short* qbf = (unsigned short*)(ws + off);    off += TB;
  unsigned short* kbf = (unsigned short*)(ws + off);    off += TB;
  unsigned short* vT  = (unsigned short*)(ws + off);    off += TB;
  unsigned short* vsT = (unsigned short*)(ws + off);    off += TB;
  unsigned short* vhT = (unsigned short*)(ws + off);    off += TB;
  unsigned short* Ob  = (unsigned short*)(ws + off);    off += TB;
  unsigned short* Osb = (unsigned short*)(ws + off);    off += TB;
  unsigned short* Ohb = (unsigned short*)(ws + off);

  // Convert the 8 weight matrices to bf16: Wq,Wk,Wv,Wvs,Wvh,Wpx,Wps,Wph.
  const int widx[8] = {6, 8, 10, 12, 14, 16, 18, 20};
  for (int i = 0; i < 8; ++i)
    cvt_bf16_kernel<<<DIMC * DIMC / 512, 256, 0, stream>>>(
        (const float*)d_in[widx[i]], Wbf + (size_t)i * DIMC * DIMC, DIMC * DIMC);

  const dim3 gg(MROWS / BM, DIMC / BN);     // 512 x 2
  const float qscale = 0.17677669529663687f;  // HEAD_DIM^-0.5

  // Input projections (A = f32 activations, B = bf16 weights).
  gemm_kernel<0><<<gg, 256, 0, stream>>>(d_in[0], Wbf + 0 * DIMC * DIMC,
                                         (const float*)d_in[7],  qbf, qscale);
  gemm_kernel<0><<<gg, 256, 0, stream>>>(d_in[1], Wbf + 1 * DIMC * DIMC,
                                         (const float*)d_in[9],  kbf, 1.0f);
  gemm_kernel<1><<<gg, 256, 0, stream>>>(d_in[2], Wbf + 2 * DIMC * DIMC,
                                         (const float*)d_in[11], vT,  1.0f);
  gemm_kernel<1><<<gg, 256, 0, stream>>>(d_in[3], Wbf + 3 * DIMC * DIMC,
                                         (const float*)d_in[13], vsT, 1.0f);
  gemm_kernel<1><<<gg, 256, 0, stream>>>(d_in[4], Wbf + 4 * DIMC * DIMC,
                                         (const float*)d_in[15], vhT, 1.0f);

  // Windowed attention over 1024 windows x 8 heads.
  attn_kernel<<<BTOT * HEADS, 128, 0, stream>>>(qbf, kbf, vT, vsT, vhT, mask,
                                                bias_table, rel_idx, Ob, Osb, Ohb);

  // Output projections straight into d_out (f32), concatenated x/scale/shift.
  float* out = (float*)d_out;
  const size_t OS = (size_t)MROWS * DIMC;
  gemm_kernel<2><<<gg, 256, 0, stream>>>(Ob,  Wbf + 5 * DIMC * DIMC,
                                         (const float*)d_in[17], out,          1.0f);
  gemm_kernel<2><<<gg, 256, 0, stream>>>(Osb, Wbf + 6 * DIMC * DIMC,
                                         (const float*)d_in[19], out + OS,     1.0f);
  gemm_kernel<2><<<gg, 256, 0, stream>>>(Ohb, Wbf + 7 * DIMC * DIMC,
                                         (const float*)d_in[21], out + 2 * OS, 1.0f);
}
